// LSTM_6811818132104
// MI455X (gfx1250) — matrix operand order repaired and checked by reading, not compiled, not run on hardware
//
#include <hip/hip_runtime.h>
#include <hip/hip_bf16.h>
#include <math.h>

// ---------------- problem constants ----------------
constexpr int Tn = 1024;   // time steps
constexpr int Bn = 64;     // batch
constexpr int In = 256;    // input features
constexpr int Hn = 256;    // hidden
constexpr int G3 = 3 * Hn; // 768 gate width

// ---------------- CDNA5 WMMA types ----------------
typedef __bf16 bf16_t;
typedef bf16_t v8bf  __attribute__((ext_vector_type(8)));
typedef bf16_t v16bf __attribute__((ext_vector_type(16)));
typedef float  v8f   __attribute__((ext_vector_type(8)));

// ---------------- TDM descriptor vector types ----------------
typedef unsigned int u32x4 __attribute__((ext_vector_type(4)));
typedef int          i32x4 __attribute__((ext_vector_type(4)));
typedef int          i32x8 __attribute__((ext_vector_type(8)));

__device__ __forceinline__ unsigned short f2bf(float x) {
    unsigned int u = __float_as_uint(x);
    unsigned int r = u + 0x7FFFu + ((u >> 16) & 1u); // round-to-nearest-even
    return (unsigned short)(r >> 16);
}

__device__ __forceinline__ float sigmoidf_fast(float x) {
    return 1.0f / (1.0f + __expf(-x));
}

// A-matrix fragment (16x32 bf16, row-major source with arbitrary row stride).
// Lane L<16: row L,  K chunks [k0, k0+8) and [k0+16, k0+24)
// Lane L>=16: row L-16, K chunks [k0+8, k0+16) and [k0+24, k0+32)
__device__ __forceinline__ v16bf load_a_frag(const unsigned short* base,
                                             int rowStride, int lane) {
    int row  = lane & 15;
    int koff = (lane >> 4) << 3; // +8 for upper half-wave
    const unsigned short* p = base + (size_t)row * rowStride + koff;
    v8bf lo = *(const v8bf*)(p);
    v8bf hi = *(const v8bf*)(p + 16);
    return __builtin_shufflevector(lo, hi, 0,1,2,3,4,5,6,7,8,9,10,11,12,13,14,15);
}

// B-matrix fragment (32x16 bf16). B = W^T where W is row-major [n][k] with
// stride rowStride, so B column n == W row n (contiguous).
// Lane L<16: col L,  K [k0, k0+8) and [k0+16, k0+24);  Lane L>=16: col L-16, K [k0+8, k0+16) and [k0+24, k0+32).
__device__ __forceinline__ v16bf load_b_frag(const unsigned short* base,
                                             int rowStride, int lane) {
    int col  = lane & 15;
    int koff = (lane >> 4) << 3; // +8 for upper half-wave
    const unsigned short* p = base + (size_t)col * rowStride + koff;
    v8bf lo = *(const v8bf*)(p);
    v8bf hi = *(const v8bf*)(p + 16);
    return __builtin_shufflevector(lo, hi, 0,1,2,3,4,5,6,7,8,9,10,11,12,13,14,15);
}

__device__ __forceinline__ v8f wmma_bf16(v16bf a, v16bf b, v8f c) {
    return __builtin_amdgcn_wmma_f32_16x16x32_bf16(
        /*neg_a=*/false, a, /*neg_b=*/false, b,
        /*c_mod=*/(short)0, c, /*reuse_a=*/false, /*reuse_b=*/false);
}

// ---------------- TDM: DMA one gx slab [64][768] f32 into LDS ----------------
// D# per CDNA5 ISA:
//  group0: count=1 | lds_addr | global_addr[56:0] | type=2
//  group1: data_size=4B; tensor_dim0=768, tensor_dim1=64;
//          tile_dim0=768, tile_dim1=64; tensor_dim0_stride=768
//  groups 2/3: zero (2-D tensor)
__device__ __forceinline__ void tdm_load_gx_slab(const float* gsrc,
                                                 unsigned ldsOff) {
    unsigned long long ga = (unsigned long long)(uintptr_t)gsrc;
    u32x4 g0;
    g0[0] = 1u;                                   // count=1 valid descriptor
    g0[1] = ldsOff;                               // lds_addr (bytes)
    g0[2] = (unsigned)(ga & 0xFFFFFFFFu);         // global_addr[31:0]
    g0[3] = (unsigned)((ga >> 32) & 0x01FFFFFFu)  // global_addr[56:32]
          | (2u << 30);                           // type=2 ("image")
    i32x8 g1;
    g1[0] = (int)(2u << 16);        // workgroup_mask=0, data_size=2 (4 bytes)
    g1[1] = (int)(768u << 16);      // tensor_dim0 low16 in [31:16]
    g1[2] = (int)(64u << 16);       // tensor_dim0 hi=0; tensor_dim1=64
    g1[3] = (int)(768u << 16);      // tensor_dim1 hi=0; tile_dim0=768
    g1[4] = 64;                     // tile_dim1=64, tile_dim2=0
    g1[5] = 768;                    // tensor_dim0_stride low32
    g1[6] = 0;                      // stride hi, tensor_dim1_stride lo
    g1[7] = 0;
    i32x4 g2; g2[0]=0; g2[1]=0; g2[2]=0; g2[3]=0;
    i32x4 g3; g3[0]=0; g3[1]=0; g3[2]=0; g3[3]=0;
#if __has_include(<hip/amd_detail/amd_gfx1250_TDM.h>)
    // amdgpu-toolchain (therock-10.0 headers): 6-arg builtin
    i32x8 g4; g4[0]=0; g4[1]=0; g4[2]=0; g4[3]=0; g4[4]=0; g4[5]=0; g4[6]=0; g4[7]=0;
    __builtin_amdgcn_tensor_load_to_lds(g0, g1, g2, g3, g4, 0);
#else
    // ROCm 7.2 (clang-22): 5-arg builtin
    __builtin_amdgcn_tensor_load_to_lds(g0, g1, g2, g3, 0);
#endif
}

__device__ __forceinline__ unsigned lds_addr_of(const void* p) {
    // LDS aperture: low 32 bits of the flat address are the LDS byte offset
    return (unsigned)(reinterpret_cast<uintptr_t>(p));
}

// ---------------- conversion kernels ----------------
__global__ void cvt_f32_to_bf16(const float* __restrict__ src,
                                unsigned short* __restrict__ dst, int n) {
    int i = blockIdx.x * blockDim.x + threadIdx.x;
    if (i < n) dst[i] = f2bf(src[i]);
}

__global__ void bias_combine(const float* __restrict__ a,
                             const float* __restrict__ b,
                             float* __restrict__ dst, int n) {
    int i = blockIdx.x * blockDim.x + threadIdx.x;
    if (i < n) dst[i] = a[i] + b[i];
}

// ---------------- input-projection GEMM ----------------
// gx[dir][t][b][g] = bias[dir][g] + sum_i Xbf[b][t][i] * Wih[dir][g][i]
__global__ __launch_bounds__(256)
void gx_gemm(const unsigned short* __restrict__ Xbf,
             const unsigned short* __restrict__ wihbf,
             const float* __restrict__ bias,
             float* __restrict__ gx) {
    const int t    = blockIdx.y;
    const int dir  = blockIdx.z;
    const int lane = threadIdx.x & 31;
    const int wave = threadIdx.x >> 5;

    const int tile = blockIdx.x * 8 + wave;   // 0..191 (4 M-tiles x 48 N-tiles)
    const int m0   = (tile & 3) * 16;         // batch rows
    const int n0   = (tile >> 2) * 16;        // gate cols

    const unsigned short* W  = wihbf + (size_t)dir * G3 * In;
    const float*          bs = bias  + (size_t)dir * G3;
    float* G = gx + ((size_t)dir * Tn + t) * Bn * G3;

    v8f acc;
    {
        float bv = bs[n0 + (lane & 15)];
        #pragma unroll
        for (int r = 0; r < 8; ++r) acc[r] = bv;
    }

    const unsigned short* Abase = Xbf + (size_t)m0 * (Tn * In) + (size_t)t * In;
    const unsigned short* Bbase = W + (size_t)n0 * In;

    #pragma unroll
    for (int k0 = 0; k0 < In; k0 += 32) {
        v16bf a = load_a_frag(Abase + k0, Tn * In, lane);
        v16bf b = load_b_frag(Bbase + k0, In, lane);
        acc = wmma_bf16(a, b, acc);
    }

    const int rbase = (lane >> 4) << 3;
    const int col   = lane & 15;
    #pragma unroll
    for (int r = 0; r < 8; ++r)
        G[(size_t)(m0 + rbase + r) * G3 + n0 + col] = acc[r];
}

// ---------------- persistent recurrent scan ----------------
// One workgroup per direction; 1024 threads = 32 waves; each wave owns two
// gate-triples so activations + cell update stay in accumulator registers.
// gx slabs are DMA'd into LDS by the Tensor Data Mover one step ahead.
__global__ __launch_bounds__(1024)
void lstm_scan(const unsigned short* __restrict__ whhbf,
               const float* __restrict__ gx,
               float* __restrict__ out,   // [T][B][2H]
               float* __restrict__ hn,    // [B][2H]
               float* __restrict__ cn) {  // [B][2H]
    const int dir  = blockIdx.x;
    const int tid  = threadIdx.x;
    const int lane = tid & 31;
    const int wave = tid >> 5;

    __shared__ float          gS[Bn * G3];      // 192 KB gx slab (TDM target)
    __shared__ unsigned short hS[Bn][Hn + 8];   // 33 KB bf16 hidden state

    const unsigned short* W = whhbf + (size_t)dir * G3 * Hn;
    const float*          G = gx + (size_t)dir * Tn * Bn * G3;

    // zero h
    {
        unsigned short* hflat = &hS[0][0];
        for (int i = tid; i < Bn * (Hn + 8); i += 1024) hflat[i] = 0;
    }

    float cstate[2][8];
    #pragma unroll
    for (int j = 0; j < 2; ++j)
        #pragma unroll
        for (int r = 0; r < 8; ++r) cstate[j][r] = 0.0f;

    const int rbase = (lane >> 4) << 3;
    const int col   = lane & 15;
    const unsigned gS_lds = lds_addr_of(&gS[0]);

    // prologue: DMA first slab, wait, publish to all waves
    if (wave == 0) {
        const int ts0 = (dir == 0) ? 0 : (Tn - 1);
        tdm_load_gx_slab(G + (size_t)ts0 * Bn * G3, gS_lds);
        __builtin_amdgcn_s_wait_tensorcnt(0);
    }
    __syncthreads();

    for (int s = 0; s < Tn; ++s) {
        const int ts = (dir == 0) ? s : (Tn - 1 - s);

        v8f acc[2][3];

        // ---- acc init: gates = gx[ts] (from LDS slab) ----
        #pragma unroll
        for (int j = 0; j < 2; ++j) {
            const int id = wave * 2 + j;
            const int m0 = (id & 3) * 16;
            const int n0 = (id >> 2) * 16;
            #pragma unroll
            for (int g = 0; g < 3; ++g) {
                const float* Gp = &gS[g * Hn + n0 + col];
                #pragma unroll
                for (int r = 0; r < 8; ++r)
                    acc[j][g][r] = Gp[(m0 + rbase + r) * G3];
            }
        }

        __syncthreads();   // all reads of gS complete -> safe for TDM overwrite

        // kick off next step's DMA; it overlaps the WMMA loop + activations
        if (wave == 0 && s + 1 < Tn) {
            const int tnext = (dir == 0) ? (s + 1) : (Tn - 2 - s);
            tdm_load_gx_slab(G + (size_t)tnext * Bn * G3, gS_lds);
        }

        // ---- WMMA: gates += h @ Whh^T (bf16 -> f32) ----
        #pragma unroll
        for (int j = 0; j < 2; ++j) {
            const int id = wave * 2 + j;
            const int m0 = (id & 3) * 16;
            const int n0 = (id >> 2) * 16;
            #pragma unroll
            for (int k0 = 0; k0 < Hn; k0 += 32) {
                v16bf a = load_a_frag(&hS[m0][0] + k0, Hn + 8, lane);
                #pragma unroll
                for (int g = 0; g < 3; ++g) {
                    v16bf b = load_b_frag(W + (size_t)(g * Hn + n0) * Hn + k0,
                                          Hn, lane);
                    acc[j][g] = wmma_bf16(a, b, acc[j][g]);
                }
            }
        }

        __syncthreads();   // all reads of hS complete

        // ---- coupled-gate activations, cell update, h write ----
        #pragma unroll
        for (int j = 0; j < 2; ++j) {
            const int id = wave * 2 + j;
            const int m0 = (id & 3) * 16;
            const int n0 = (id >> 2) * 16;
            #pragma unroll
            for (int r = 0; r < 8; ++r) {
                float ig = sigmoidf_fast(acc[j][0][r]);
                float cg = tanhf(acc[j][1][r]);
                float og = sigmoidf_fast(acc[j][2][r]);
                float cv = (1.0f - ig) * cstate[j][r] + ig * cg;
                cstate[j][r] = cv;
                float hv = og * tanhf(cv);

                const int b  = m0 + rbase + r;
                const int hh = n0 + col;
                hS[b][hh] = f2bf(hv);
                out[((size_t)ts * Bn + b) * (2 * Hn) + dir * Hn + hh] = hv;
                if (s == Tn - 1) {
                    hn[(size_t)b * (2 * Hn) + dir * Hn + hh] = hv;
                    cn[(size_t)b * (2 * Hn) + dir * Hn + hh] = cv;
                }
            }
        }

        // DMA for next slab must have landed before anyone reads gS again
        if (wave == 0 && s + 1 < Tn) {
            __builtin_amdgcn_s_wait_tensorcnt(0);
        }
        __syncthreads();   // h + next gx slab visible to all waves
    }
}

// ---------------- host-side launcher ----------------
extern "C" void kernel_launch(void* const* d_in, const int* in_sizes, int n_in,
                              void* d_out, int out_size, void* d_ws, size_t ws_size,
                              hipStream_t stream) {
    (void)in_sizes; (void)n_in; (void)out_size; (void)ws_size;

    const float* X     = (const float*)d_in[0];
    const float* wih_f = (const float*)d_in[1];
    const float* whh_f = (const float*)d_in[2];
    const float* bih_f = (const float*)d_in[3];
    const float* bhh_f = (const float*)d_in[4];
    const float* wih_b = (const float*)d_in[5];
    const float* whh_b = (const float*)d_in[6];
    const float* bih_b = (const float*)d_in[7];
    const float* bhh_b = (const float*)d_in[8];

    char* ws = (char*)d_ws;
    const size_t xbf_bytes = (size_t)Bn * Tn * In * 2;        // 32 MB
    const size_t w_bytes   = (size_t)G3 * In * 2;             // 384 KB each
    unsigned short* Xbf   = (unsigned short*)(ws);
    unsigned short* wihbf = (unsigned short*)(ws + xbf_bytes);
    unsigned short* whhbf = (unsigned short*)(ws + xbf_bytes + 2 * w_bytes);
    float*          bias  = (float*)(ws + xbf_bytes + 4 * w_bytes);
    float*          gxbuf = (float*)(ws + xbf_bytes + 4 * w_bytes
                                        + 2 * G3 * sizeof(float));

    // 1) convert inputs to bf16, combine biases
    {
        const int nX = Bn * Tn * In;
        cvt_f32_to_bf16<<<(nX + 255) / 256, 256, 0, stream>>>(X, Xbf, nX);
        const int nW = G3 * In;
        cvt_f32_to_bf16<<<(nW + 255) / 256, 256, 0, stream>>>(wih_f, wihbf, nW);
        cvt_f32_to_bf16<<<(nW + 255) / 256, 256, 0, stream>>>(wih_b, wihbf + nW, nW);
        cvt_f32_to_bf16<<<(nW + 255) / 256, 256, 0, stream>>>(whh_f, whhbf, nW);
        cvt_f32_to_bf16<<<(nW + 255) / 256, 256, 0, stream>>>(whh_b, whhbf + nW, nW);
        bias_combine<<<(G3 + 255) / 256, 256, 0, stream>>>(bih_f, bhh_f, bias, G3);
        bias_combine<<<(G3 + 255) / 256, 256, 0, stream>>>(bih_b, bhh_b, bias + G3, G3);
    }

    // 2) big input-projection GEMM for both directions
    {
        dim3 grid(24, Tn, 2);
        gx_gemm<<<grid, 256, 0, stream>>>(Xbf, wihbf, bias, gxbuf);
    }

    // 3) persistent recurrent scan, one workgroup per direction
    {
        float* out = (float*)d_out;
        float* hn  = out + (size_t)Tn * Bn * 2 * Hn;
        float* cn  = hn + (size_t)Bn * 2 * Hn;
        lstm_scan<<<2, 1024, 0, stream>>>(whhbf, gxbuf, out, hn, cn);
    }
}
